// MatrixLSTMCell_65352222376500
// MI455X (gfx1250) — matrix operand order repaired and checked by reading, not compiled, not run on hardware
//
#include <hip/hip_runtime.h>
#include <hip/hip_bf16.h>

typedef __attribute__((ext_vector_type(16))) __bf16 v16bf;
typedef __attribute__((ext_vector_type(8)))  __bf16 v8bf;
typedef __attribute__((ext_vector_type(8)))  float  v8f;
typedef __attribute__((ext_vector_type(4)))  unsigned int v4u;
typedef __attribute__((ext_vector_type(8)))  unsigned int v8u;

#define B_    256
#define S_    64
#define DIM_  1024
#define NH_   8
#define DH_   128
#define KSCALE 0.08838834764831845f   // 1/sqrt(128)

static __device__ __forceinline__ v8f wmma_bf16(v16bf a, v16bf b, v8f c) {
  return __builtin_amdgcn_wmma_f32_16x16x32_bf16(false, a, false, b, (short)0, c, false, false);
}

// A matrix element e (0..15) of the 16x32 bf16 tile maps to k = (e<8 ? e : 8+e) + laneHi*8.
// rowptr points at the row (contiguous k); loads are two 16B chunks.
static __device__ __forceinline__ v16bf load_a_bf16(const __bf16* rowptr, int laneHi, int kbase) {
  const __bf16* p = rowptr + kbase + laneHi * 8;
  v8bf lo = *(const v8bf*)(p);
  v8bf hi = *(const v8bf*)(p + 16);
  return __builtin_shufflevector(lo, hi, 0,1,2,3,4,5,6,7,8,9,10,11,12,13,14,15);
}

// B matrix: lane%16 = n column; element e maps to k = (e<8 ? e : 8+e) + laneHi*8, row stride ld.
static __device__ __forceinline__ v16bf load_b_bf16(const __bf16* p, int ld, int laneHi) {
  v16bf r;
#pragma unroll
  for (int e = 0; e < 16; ++e) r[e] = p[((e < 8 ? e : 8 + e) + laneHi * 8) * ld];
  return r;
}

// TDM: DMA a contiguous run of `n` bf16 elements from global to LDS offset 0.
// D# per CDNA5 ISA ch.8: group0={count,lds_addr,global_addr,type=2},
// group1={data_size=2B, tensor_dim0=n, tile_dim0=n, dim0_stride=n}, groups2/3 = 1-extent dims.
static __device__ __forceinline__ void tdm_load_lds0(const void* gptr, unsigned n) {
  unsigned long long ga = (unsigned long long)gptr;
  v4u g0;
  g0.x = 1u;                                                  // count=1 (user mode)
  g0.y = 0u;                                                  // lds_addr = 0
  g0.z = (unsigned)(ga & 0xffffffffull);                      // global_addr[31:0]
  g0.w = (unsigned)((ga >> 32) & 0x1ffffffull) | (2u << 30);  // addr[56:32] | type=2
  v8u g1;
  g1[0] = (1u << 16);                        // data_size=1 -> 2 bytes
  g1[1] = (n & 0xffffu) << 16;               // tensor_dim0[15:0]  (bits 63:48)
  g1[2] = (n >> 16) | (1u << 16);            // tensor_dim0[31:16] | tensor_dim1=1
  g1[3] = (n & 0xffffu) << 16;               // tile_dim0 = n      (bits 127:112)
  g1[4] = 1u;                                // tile_dim1 = 1
  g1[5] = n;                                 // tensor_dim0_stride lo32
  g1[6] = 0u;
  g1[7] = 0u;
  v4u g2; g2.x = 1u; g2.y = 1u; g2.z = 0u; g2.w = 0u;   // tensor_dim2=1, tensor_dim3=1
  v4u g3; g3.x = 0u; g3.y = (1u << 16); g3.z = 0u; g3.w = 0u; // tensor_dim4=1
  asm volatile("tensor_load_to_lds %0, %1, %2, %3"
               :
               : "s"(g0), "s"(g1), "s"(g2), "s"(g3)
               : "memory");
}

// ---------------- prep: q,v -> bf16 (b,h,s,c); k -> bf16 transposed (b,h,c,t) * KSCALE
__global__ void prep_qkv(const float* __restrict__ q, const float* __restrict__ k,
                         const float* __restrict__ v,
                         __bf16* __restrict__ qhb, __bf16* __restrict__ khbT,
                         __bf16* __restrict__ vhb) {
  __shared__ float lds_k[64 * 130];
  int bh = blockIdx.x;            // b*8+h
  int b = bh >> 3, h = bh & 7;
  int tid = threadIdx.x;
  for (int i = tid; i < 64 * 128; i += 256) {
    int s = i >> 7, c = i & 127;
    long gi = ((long)(b * 64 + s)) * 1024 + h * 128 + c;
    long oi = (long)bh * 8192 + i;           // ((b*8+h)*64+s)*128+c
    qhb[oi] = (__bf16)q[gi];
    vhb[oi] = (__bf16)v[gi];
    lds_k[s * 130 + c] = k[gi] * KSCALE;
  }
  __syncthreads();
  for (int i = tid; i < 64 * 128; i += 256) {
    int c = i >> 6, s = i & 63;
    khbT[(long)bh * 8192 + i] = (__bf16)lds_k[s * 130 + c];  // bh*8192 + c*64 + t
  }
}

// ---------------- prep: pe_t[h][s][c][t] = bf16(rpe[hops[s][t]][h*128+c])
__global__ void prep_pe(const float* __restrict__ rpe, const int* __restrict__ hops,
                        __bf16* __restrict__ pe_t) {
  __shared__ float lds[64 * 130];   // [t][c]
  int hs = blockIdx.x;              // h*64+s
  int h = hs >> 6, s = hs & 63;
  int tid = threadIdx.x;
  for (int i = tid; i < 64 * 128; i += 256) {
    int t = i >> 7, c = i & 127;
    int hop = hops[s * 64 + t];
    lds[t * 130 + c] = rpe[(long)hop * 1024 + h * 128 + c];
  }
  __syncthreads();
  for (int i = tid; i < 64 * 128; i += 256) {
    int c = i >> 6, t = i & 63;
    pe_t[(long)hs * 8192 + i] = (__bf16)lds[t * 130 + c];   // hs*8192 + c*64 + t
  }
}

// ---------------- gates: (B*S,3072)@(3072,16) via WMMA; outputs ig/fg [b][h][s]
__global__ void gates_kernel(const float* __restrict__ q, const float* __restrict__ k,
                             const float* __restrict__ v,
                             const float* __restrict__ igw, const float* __restrict__ igb,
                             const float* __restrict__ fgw, const float* __restrict__ fgb,
                             float* __restrict__ ig_ws, float* __restrict__ fg_ws) {
  __shared__ __align__(16) __bf16 wT[1024 * 16];  // staged W^T chunk: [c][o], 32KB
  int b = blockIdx.x, tid = threadIdx.x;
  int wave = tid >> 5, lane = tid & 31;
  int laneHi = lane >> 4, lm = lane & 15;
  int s = wave * 16 + lm;                  // A row for loads
  v8f acc = {};
  for (int stage = 0; stage < 3; ++stage) {
    __syncthreads();
    for (int i = tid; i < 1024 * 16; i += 128) {
      int o = i >> 10, c = i & 1023;
      float wv = (o < 8) ? igw[o * 3072 + stage * 1024 + c]
                         : fgw[(o - 8) * 3072 + stage * 1024 + c];
      wT[c * 16 + o] = (__bf16)wv;
    }
    __syncthreads();
    const float* arr = (stage == 0) ? q : ((stage == 1) ? k : v);
    const float* rowbase = arr + ((long)(b * 64 + s)) * 1024;
#pragma unroll 4
    for (int kc = 0; kc < 32; ++kc) {
      int cc = kc * 32;
      const float* rp = rowbase + cc + laneHi * 8;
      float4 f0 = *(const float4*)(rp);
      float4 f1 = *(const float4*)(rp + 4);
      float4 f2 = *(const float4*)(rp + 16);
      float4 f3 = *(const float4*)(rp + 20);
      v16bf a;
      a[0]=(__bf16)f0.x; a[1]=(__bf16)f0.y; a[2]=(__bf16)f0.z; a[3]=(__bf16)f0.w;
      a[4]=(__bf16)f1.x; a[5]=(__bf16)f1.y; a[6]=(__bf16)f1.z; a[7]=(__bf16)f1.w;
      a[8]=(__bf16)f2.x; a[9]=(__bf16)f2.y; a[10]=(__bf16)f2.z; a[11]=(__bf16)f2.w;
      a[12]=(__bf16)f3.x; a[13]=(__bf16)f3.y; a[14]=(__bf16)f3.z; a[15]=(__bf16)f3.w;
      v16bf bt = load_b_bf16(&wT[cc * 16 + lm], 16, laneHi);
      acc = wmma_bf16(a, bt, acc);
    }
  }
  int o = lm;
  float bias = (o < 8) ? igb[o] : fgb[o - 8];
  float* dst = (o < 8) ? ig_ws : fg_ws;
  int oo = (o < 8) ? o : o - 8;
#pragma unroll
  for (int r = 0; r < 8; ++r) {
    int srow = wave * 16 + r + laneHi * 8;
    dst[b * 512 + oo * 64 + srow] = acc[r] + bias;
  }
}

// ---------------- khop: per (h,s): (B x 64 x 128) GEMM over batch; pe tile via TDM
__global__ void khop_kernel(const __bf16* __restrict__ qhb, const __bf16* __restrict__ pe_t,
                            float* __restrict__ khop) {
  __shared__ __align__(16) __bf16 lpe[8192];  // [c][t] 128x64 -- ONLY LDS object (offset 0)
  int hs = blockIdx.x;   // h*64+s
  int h = hs >> 6, s = hs & 63;
  int tid = threadIdx.x;
  if (tid < 32) {
    // Tensor Data Mover: DMA 16KB pe tile global->LDS; wave waits on TENSORcnt.
    tdm_load_lds0(pe_t + (long)hs * 8192, 8192u);
    __builtin_amdgcn_s_wait_tensorcnt(0);
  }
  __syncthreads();
  int wave = tid >> 5, lane = tid & 31, laneHi = lane >> 4, lm = lane & 15;
  v8f acc[2][4];
#pragma unroll
  for (int mt = 0; mt < 2; ++mt)
#pragma unroll
    for (int nt = 0; nt < 4; ++nt) acc[mt][nt] = (v8f){};
#pragma unroll
  for (int kc = 0; kc < 4; ++kc) {
    int c0 = kc * 32;
    v16bf a[2];
#pragma unroll
    for (int mt = 0; mt < 2; ++mt) {
      int bb = wave * 32 + mt * 16 + lm;
      const __bf16* rowp = qhb + (((long)bb * 8 + h) * 64 + s) * 128;
      a[mt] = load_a_bf16(rowp, laneHi, c0);
    }
#pragma unroll
    for (int nt = 0; nt < 4; ++nt) {
      v16bf bt = load_b_bf16(&lpe[c0 * 64 + nt * 16 + lm], 64, laneHi);
#pragma unroll
      for (int mt = 0; mt < 2; ++mt) acc[mt][nt] = wmma_bf16(a[mt], bt, acc[mt][nt]);
    }
  }
#pragma unroll
  for (int mt = 0; mt < 2; ++mt)
#pragma unroll
    for (int nt = 0; nt < 4; ++nt)
#pragma unroll
      for (int r = 0; r < 8; ++r) {
        int bb = wave * 32 + mt * 16 + r + laneHi * 8;
        int t = nt * 16 + lm;
        khop[(((long)bb * 8 + h) * 64 + s) * 64 + t] = acc[mt][nt][r];
      }
}

// ---------------- attention + gating + groupnorm per (b,h)
__global__ void attn_kernel(const __bf16* __restrict__ qhb, const __bf16* __restrict__ khbT,
                            const __bf16* __restrict__ vhb, const float* __restrict__ khop,
                            const float* __restrict__ ig_ws, const float* __restrict__ fg_ws,
                            const float* __restrict__ ln_w, const float* __restrict__ ln_b,
                            float* __restrict__ out) {
  __shared__ __align__(16) float  lds_C[64 * 65];
  __shared__ __align__(16) __bf16 lds_A[64 * 72];
  __shared__ __align__(16) float  lds_h[64 * 130];
  __shared__ float csum[65];
  __shared__ float igs[64];
  __shared__ float muA[64], rstdA[64];
  int bh = blockIdx.x;  // b*8+h
  int b = bh >> 3, h = bh & 7;
  int tid = threadIdx.x;
  // pull the per-(b,h) khop tile toward cache before phase 1 consumes it
  __builtin_prefetch(&khop[(long)bh * 4096 + tid * 16], 0, 3);
  // phase 0: gates -> LDS (stable log-sigmoid cumsum, serial on t0; tiny)
  if (tid < 64) igs[tid] = ig_ws[b * 512 + h * 64 + tid];
  if (tid == 0) {
    float c = 0.f; csum[0] = 0.f;
    for (int i = 0; i < 64; ++i) {
      float x = fg_ws[b * 512 + h * 64 + i];
      float lf = fminf(x, 0.f) - log1pf(expf(-fabsf(x)));
      c += lf; csum[i + 1] = c;
    }
  }
  int wave = tid >> 5, lane = tid & 31, laneHi = lane >> 4, lm = lane & 15;
  const __bf16* qbase = qhb + (long)bh * 8192;
  const __bf16* ktbase = khbT + (long)bh * 8192;
  // phase 1: qk*scale + khop -> lds_C
  int mi = wave >> 1;
#pragma unroll
  for (int nj = 0; nj < 2; ++nj) {
    int ni = (wave & 1) * 2 + nj;
    v8f acc = {};
#pragma unroll
    for (int kc = 0; kc < 4; ++kc) {
      int c0 = kc * 32;
      v16bf a = load_a_bf16(qbase + (mi * 16 + lm) * 128, laneHi, c0);
      v16bf bt = load_b_bf16(ktbase + c0 * 64 + ni * 16 + lm, 64, laneHi);
      acc = wmma_bf16(a, bt, acc);
    }
#pragma unroll
    for (int r = 0; r < 8; ++r) {
      int srow = mi * 16 + r + laneHi * 8;
      int t = ni * 16 + lm;
      lds_C[srow * 65 + t] = acc[r] + khop[((long)bh * 64 + srow) * 64 + t];
    }
  }
  __syncthreads();
  // phase 2: row-wise gating / stabilization / normalization -> lds_A (bf16)
  if (tid < 64) {
    int s = tid;
    float cs = csum[s + 1];
    float maxd = -3.4e38f;
    for (int t = 0; t <= s; ++t) {
      float ld = cs - csum[t + 1] + igs[t];
      maxd = fmaxf(maxd, ld);
    }
    float sum = 0.f;
    for (int t = 0; t <= s; ++t) {
      float ld = cs - csum[t + 1] + igs[t];
      float cv = lds_C[s * 65 + t] * expf(ld - maxd);
      lds_C[s * 65 + t] = cv;
      sum += cv;
    }
    float denom = fmaxf(fabsf(sum), expf(-maxd)) + 1e-6f;
    float inv = 1.f / denom;
    for (int t = 0; t < 64; ++t) {
      float cn = (t <= s) ? lds_C[s * 65 + t] * inv : 0.f;
      lds_A[s * 72 + t] = (__bf16)cn;
    }
  }
  __syncthreads();
  // phase 3: h = Cn @ V
  {
    const __bf16* vbase = vhb + (long)bh * 8192;
    int mi3 = wave >> 1;
#pragma unroll
    for (int nj = 0; nj < 4; ++nj) {
      int ni = (wave & 1) * 4 + nj;
      v8f acc = {};
#pragma unroll
      for (int kc = 0; kc < 2; ++kc) {
        int c0 = kc * 32;
        v16bf a = load_a_bf16(&lds_A[(mi3 * 16 + lm) * 72], laneHi, c0);
        v16bf bt = load_b_bf16(vbase + c0 * 128 + ni * 16 + lm, 128, laneHi);
        acc = wmma_bf16(a, bt, acc);
      }
#pragma unroll
      for (int r = 0; r < 8; ++r) {
        int srow = mi3 * 16 + r + laneHi * 8;
        int d = ni * 16 + lm;
        lds_h[srow * 130 + d] = acc[r];
      }
    }
  }
  __syncthreads();
  // phase 4: per-head groupnorm stats
  if (tid < 64) {
    int s = tid;
    float m = 0.f;
    for (int d = 0; d < 128; ++d) m += lds_h[s * 130 + d];
    m *= (1.f / 128.f);
    float vv = 0.f;
    for (int d = 0; d < 128; ++d) { float x = lds_h[s * 130 + d] - m; vv += x * x; }
    vv *= (1.f / 128.f);
    muA[s] = m; rstdA[s] = rsqrtf(vv + 1e-5f);
  }
  __syncthreads();
  for (int i = tid; i < 8192; i += 256) {
    int s = i >> 7, d = i & 127;
    float w = 1.f + ln_w[h * 128 + d];
    float bb = ln_b[h * 128 + d];
    float val = (lds_h[s * 130 + d] - muA[s]) * rstdA[s] * w + bb;
    out[((long)(b * 64 + s)) * 1024 + h * 128 + d] = val;
  }
}

extern "C" void kernel_launch(void* const* d_in, const int* in_sizes, int n_in,
                              void* d_out, int out_size, void* d_ws, size_t ws_size,
                              hipStream_t stream) {
  const float* q   = (const float*)d_in[0];
  const float* k   = (const float*)d_in[1];
  const float* v   = (const float*)d_in[2];
  const float* igw = (const float*)d_in[3];
  const float* igb = (const float*)d_in[4];
  const float* fgw = (const float*)d_in[5];
  const float* fgb = (const float*)d_in[6];
  const float* rpe = (const float*)d_in[7];
  const float* lnw = (const float*)d_in[8];
  const float* lnb = (const float*)d_in[9];
  const int*  hops = (const int*)d_in[10];
  float* out = (float*)d_out;

  char* ws = (char*)d_ws;
  const size_t QKV_BF = (size_t)B_ * NH_ * S_ * DH_ * sizeof(__bf16);  // 33.5MB each
  __bf16* qhb  = (__bf16*)(ws);
  __bf16* khbT = (__bf16*)(ws + QKV_BF);
  __bf16* vhb  = (__bf16*)(ws + 2 * QKV_BF);
  __bf16* pe_t = (__bf16*)(ws + 3 * QKV_BF);                            // 8.4MB
  float*  khop = (float*)(ws + 3 * QKV_BF + (size_t)NH_ * S_ * DH_ * S_ * sizeof(__bf16));
  float*  igws = (float*)((char*)khop + (size_t)B_ * NH_ * S_ * S_ * sizeof(float));
  float*  fgws = igws + (size_t)B_ * NH_ * S_;

  prep_qkv<<<B_ * NH_, 256, 0, stream>>>(q, k, v, qhb, khbT, vhb);
  prep_pe<<<NH_ * S_, 256, 0, stream>>>(rpe, hops, pe_t);
  gates_kernel<<<B_, 128, 0, stream>>>(q, k, v, igw, igb, fgw, fgb, igws, fgws);
  khop_kernel<<<NH_ * S_, 256, 0, stream>>>(qhb, pe_t, khop);
  attn_kernel<<<B_ * NH_, 256, 0, stream>>>(qhb, khbT, vhb, khop, igws, fgws, lnw, lnb, out);
}
